// MultiHeadAttention_34093450396181
// MI455X (gfx1250) — matrix operand order repaired and checked by reading, not compiled, not run on hardware
//
#include <hip/hip_runtime.h>
#include <cstdint>
#include <cstddef>

typedef __bf16 bf16;
typedef __attribute__((ext_vector_type(16))) __bf16 bf16x16;
typedef __attribute__((ext_vector_type(8)))  __bf16 bf16x8;
typedef __attribute__((ext_vector_type(8)))  float  f32x8;

#define DEV_INLINE __device__ __forceinline__

constexpr int Bb = 4, Ts = 2048, Cc = 1024, Hh = 16, Dd = 64;
constexpr int Mrows = Bb * Ts;   // 8192
constexpr int N3    = 3 * Cc;    // 3072

DEV_INLINE bf16x16 frag_cat(bf16x8 lo, bf16x8 hi) {
    bf16x16 r;
#pragma unroll
    for (int e = 0; e < 8; ++e) { r[e] = lo[e]; r[e + 8] = hi[e]; }
    return r;
}

// 16-lane max butterfly via DPP (XOR15, XOR7, XOR2, XOR1 span the 16-lane group).
// Stays within DPP16 rows => matches the WMMA C-layout lane groups exactly.
DEV_INLINE float dpp_max16(float x) {
    int t;
    t = __builtin_amdgcn_update_dpp(0, __float_as_int(x), 0x140, 0xF, 0xF, true); // ROW_MIRROR
    x = fmaxf(x, __int_as_float(t));
    t = __builtin_amdgcn_update_dpp(0, __float_as_int(x), 0x141, 0xF, 0xF, true); // ROW_HALF_MIRROR
    x = fmaxf(x, __int_as_float(t));
    t = __builtin_amdgcn_update_dpp(0, __float_as_int(x), 0x4E, 0xF, 0xF, true);  // quad_perm [2,3,0,1]
    x = fmaxf(x, __int_as_float(t));
    t = __builtin_amdgcn_update_dpp(0, __float_as_int(x), 0xB1, 0xF, 0xF, true);  // quad_perm [1,0,3,2]
    x = fmaxf(x, __int_as_float(t));
    return x;
}

// ---------------- f32 -> bf16 conversion ----------------
__global__ void cvt_bf16_kernel(const float* __restrict__ in, bf16* __restrict__ out, int n) {
    int i = blockIdx.x * blockDim.x + threadIdx.x;
    if (i < n) out[i] = (bf16)in[i];
}

// in: [K][N] f32 row-major  ->  out: [N][K] bf16 row-major
__global__ void cvt_transpose_kernel(const float* __restrict__ in, bf16* __restrict__ out,
                                     int K, int N) {
    int i = blockIdx.x * blockDim.x + threadIdx.x;
    if (i >= K * N) return;
    int n = i % N, k = i / N;
    out[(size_t)n * K + k] = (bf16)in[i];
}

// ---------------- bf16 GEMM:  C[M,N] = A[M,K] * Bt[N,K]^T ----------------
// WG = 256 threads (8 waves as 4x2), tile 128(M) x 128(N), wave tile 32x64,
// K-step 32. Double-buffered LDS tiles filled with CDNA5 async global->LDS
// copies; one barrier per K-step, next tile's copy overlapped with WMMA.
template <typename OutT>
__global__ __launch_bounds__(256) void gemm_bf16_kernel(
    const bf16* __restrict__ A, const bf16* __restrict__ Bt,
    OutT* __restrict__ C, int M, int N, int K)
{
    __shared__ __align__(16) bf16 lA[2][128 * 32];  // 2 x 8 KB
    __shared__ __align__(16) bf16 lB[2][128 * 32];  // 2 x 8 KB

    const int tid  = threadIdx.x;
    const int lane = tid & 31;
    const int wave = tid >> 5;
    const int mw = wave >> 1, nw = wave & 1;
    const int blockM = blockIdx.y * 128;
    const int blockN = blockIdx.x * 128;

    f32x8 acc[2][4] = {};

    const unsigned ldsA[2] = { (unsigned)(size_t)(&lA[0][0]), (unsigned)(size_t)(&lA[1][0]) };
    const unsigned ldsB[2] = { (unsigned)(size_t)(&lB[0][0]), (unsigned)(size_t)(&lB[1][0]) };
    const unsigned long long aBase = (unsigned long long)A;
    const unsigned long long bBase = (unsigned long long)Bt;

    auto stage = [&](int ks, int buf) {
        // A tile: 128 rows x 32 cols bf16 = 512 x 16B chunks (2 per thread)
#pragma unroll
        for (int it = 0; it < 2; ++it) {
            int c   = tid + it * 256;
            int row = c >> 2;                       // 4 chunks per 64B row
            unsigned gb = ((unsigned)(blockM + row) * (unsigned)K + (unsigned)ks) * 2u
                          + (unsigned)((c & 3) * 16);
            unsigned lo = ldsA[buf] + (unsigned)c * 16u;
            asm volatile("global_load_async_to_lds_b128 %0, %1, %2"
                         :: "v"(lo), "v"(gb), "s"(aBase) : "memory");
        }
        // B tile: 128 rows x 32 cols bf16 = 512 x 16B chunks (2 per thread)
#pragma unroll
        for (int it = 0; it < 2; ++it) {
            int c   = tid + it * 256;
            int row = c >> 2;
            unsigned gb = ((unsigned)(blockN + row) * (unsigned)K + (unsigned)ks) * 2u
                          + (unsigned)((c & 3) * 16);
            unsigned lo = ldsB[buf] + (unsigned)c * 16u;
            asm volatile("global_load_async_to_lds_b128 %0, %1, %2"
                         :: "v"(lo), "v"(gb), "s"(bBase) : "memory");
        }
    };

    stage(0, 0);

    // A fragment (16x32 bf16): lanes 0-15 hold K 0..7 & 16..23, lanes 16-31 hold K 8..15 & 24..31
    const int aoff = (lane & 16) ? 16 : 0;
    // B fragment (32x16 bf16): lanes 0-15 hold K 0..7 & 16..23, lanes 16-31 hold K 8..15 & 24..31 (per column)
    const int boff = (lane & 16) ? 16 : 0;

    int buf = 0;
    for (int ks = 0; ks < K; ks += 32, buf ^= 1) {
        asm volatile("s_wait_asynccnt 0x0" ::: "memory");  // current buffer arrived
        __syncthreads();                                    // everyone done with other buffer
        if (ks + 32 < K) stage(ks + 32, buf ^ 1);           // prefetch next tile (overlapped)

        bf16x16 af[2], bfr[4];
#pragma unroll
        for (int i = 0; i < 2; ++i) {
            const char* base = (const char*)&lA[buf][(mw * 32 + i * 16 + (lane & 15)) * 32];
            af[i] = frag_cat(*(const bf16x8*)(base + aoff),
                             *(const bf16x8*)(base + aoff + 32));
        }
#pragma unroll
        for (int j = 0; j < 4; ++j) {
            const char* base = (const char*)&lB[buf][(nw * 64 + j * 16 + (lane & 15)) * 32];
            bfr[j] = frag_cat(*(const bf16x8*)(base + boff),
                              *(const bf16x8*)(base + boff + 32));
        }
#pragma unroll
        for (int i = 0; i < 2; ++i)
#pragma unroll
            for (int j = 0; j < 4; ++j)
                acc[i][j] = __builtin_amdgcn_wmma_f32_16x16x32_bf16(
                    false, af[i], false, bfr[j], (short)0, acc[i][j], false, false);
    }

    // Epilogue: C layout — VGPR r: lanes0-15 (M=r, N=lane), lanes16-31 (M=8+r, N=lane-16)
#pragma unroll
    for (int i = 0; i < 2; ++i)
#pragma unroll
        for (int j = 0; j < 4; ++j)
#pragma unroll
            for (int r = 0; r < 8; ++r) {
                int row = blockM + mw * 32 + i * 16 + r + ((lane & 16) ? 8 : 0);
                int col = blockN + nw * 64 + j * 16 + (lane & 15);
                C[(size_t)row * (size_t)N + col] = (OutT)acc[i][j][r];
            }
}

// ---------------- RoPE + layout change ----------------
// qkv bf16 [B*T, 3C] -> qr,kr bf16 [B,H,T,D] (rotated), vt bf16 [B,H,D,T]
__global__ void rope_kernel(const bf16* __restrict__ qkv,
                            bf16* __restrict__ qr, bf16* __restrict__ kr,
                            bf16* __restrict__ vt)
{
    int idx = blockIdx.x * blockDim.x + threadIdx.x;  // [bh][t][i], i = pair index 0..31
    int i  = idx & 31;
    int t  = (idx >> 5) & (Ts - 1);
    int bh = idx >> 16;
    if (bh >= Bb * Hh) return;
    int b = bh >> 4, h = bh & 15;

    size_t rowq = (size_t)(b * Ts + t) * N3;
    int colq = h * Dd + 2 * i;
    float q1 = (float)qkv[rowq + colq];
    float q2 = (float)qkv[rowq + colq + 1];
    float k1 = (float)qkv[rowq + Cc + colq];
    float k2 = (float)qkv[rowq + Cc + colq + 1];
    float v1 = (float)qkv[rowq + 2 * Cc + colq];
    float v2 = (float)qkv[rowq + 2 * Cc + colq + 1];

    // inv_freq = 10000^(-2i/64) = e^(-2i * ln(10000)/64)
    float inv = __expf(-(float)(2 * i) * (9.210340371976184f / 64.0f));
    float ang = (float)t * inv;
    float sn, cs;
    __sincosf(ang, &sn, &cs);

    size_t qo = ((size_t)bh * Ts + t) * Dd + 2 * i;
    qr[qo]     = (bf16)(q1 * cs - q2 * sn);
    qr[qo + 1] = (bf16)(q2 * cs + q1 * sn);
    kr[qo]     = (bf16)(k1 * cs - k2 * sn);
    kr[qo + 1] = (bf16)(k2 * cs + k1 * sn);

    size_t vo = ((size_t)bh * Dd + 2 * i) * Ts + t;
    vt[vo]      = (bf16)v1;
    vt[vo + Ts] = (bf16)v2;
}

// ---------------- Flash attention ----------------
// grid (B*H, T/64), 128 threads = 4 waves, each wave owns 16 q rows.
// 64-key kv blocks staged into double-buffered LDS with async copies and
// shared by all 4 waves. Softmax row-max uses a DPP butterfly (pure VALU);
// softmax row-sum is computed by WMMA against an all-ones B fragment, which
// lands the sums pre-broadcast across each 16-lane row group.
__global__ __launch_bounds__(128) void attn_kernel(
    const bf16* __restrict__ qr, const bf16* __restrict__ kr,
    const bf16* __restrict__ vt, bf16* __restrict__ out)
{
    __shared__ __align__(16) bf16 lK[2][64 * 64];  // 2 x 8 KB : [key][dim]
    __shared__ __align__(16) bf16 lV[2][64 * 64];  // 2 x 8 KB : [dim][key]
    __shared__ __align__(16) bf16 pb[4][16 * 64];  // per-wave P bounce (2 KB each)

    const int tid  = threadIdx.x;
    const int bh   = blockIdx.x;
    const int qb   = blockIdx.y * 64;
    const int lane = tid & 31;
    const int wave = tid >> 5;
    const int qw   = qb + wave * 16;
    const int b = bh >> 4, h = bh & 15;

    const unsigned ldsK[2] = { (unsigned)(size_t)(&lK[0][0]), (unsigned)(size_t)(&lK[1][0]) };
    const unsigned ldsV[2] = { (unsigned)(size_t)(&lV[0][0]), (unsigned)(size_t)(&lV[1][0]) };
    const unsigned long long kBase = (unsigned long long)kr;
    const unsigned long long vBase = (unsigned long long)vt;

    auto stage = [&](int kv, int bufi) {
        // K tile: keys kv..kv+63, all 64 dims -> contiguous 8KB in kr
        unsigned kgb = ((unsigned)(bh * Ts + kv)) * (unsigned)(Dd * 2);
#pragma unroll
        for (int it = 0; it < 4; ++it) {
            int c = tid + it * 128;                 // 512 x 16B chunks
            unsigned lo = ldsK[bufi] + (unsigned)c * 16u;
            unsigned gb = kgb + (unsigned)c * 16u;
            asm volatile("global_load_async_to_lds_b128 %0, %1, %2"
                         :: "v"(lo), "v"(gb), "s"(kBase) : "memory");
        }
        // V tile: 64 dim-rows x 64 keys (128B rows, global row stride Ts*2)
#pragma unroll
        for (int it = 0; it < 4; ++it) {
            int c   = tid + it * 128;
            int row = c >> 3;                       // 8 chunks per 128B row
            unsigned lo = ldsV[bufi] + (unsigned)c * 16u;
            unsigned gb = ((unsigned)(bh * Dd + row) * (unsigned)Ts + (unsigned)kv) * 2u
                          + (unsigned)((c & 7) * 16);
            asm volatile("global_load_async_to_lds_b128 %0, %1, %2"
                         :: "v"(lo), "v"(gb), "s"(vBase) : "memory");
        }
    };

    const int ao = (lane & 16) ? 16 : 0;  // A-fragment byte select
    const int bo = (lane & 16) ? 16 : 0;  // B-fragment byte select

    // q fragments: dims 0..31 and 32..63 (loaded once, straight from global)
    const char* qbase = (const char*)(qr + ((size_t)bh * Ts + (qw + (lane & 15))) * (size_t)Dd);
    bf16x16 qf[2];
#pragma unroll
    for (int c = 0; c < 2; ++c)
        qf[c] = frag_cat(*(const bf16x8*)(qbase + c * 64 + ao),
                         *(const bf16x8*)(qbase + c * 64 + ao + 32));

    // all-ones B fragment for WMMA row-sum
    bf16x16 onesf;
#pragma unroll
    for (int e = 0; e < 16; ++e) onesf[e] = (bf16)1.0f;

    float m_r[8], l_r[8];
#pragma unroll
    for (int r = 0; r < 8; ++r) { m_r[r] = -3.0e38f; l_r[r] = 0.0f; }
    f32x8 acc[4] = {};

    const int nb = qb / 64 + 1;   // uniform over WG: 64-key blocks covering keys <= qb+63
    stage(0, 0);

    int bufi = 0;
    for (int j = 0; j < nb; ++j, bufi ^= 1) {
        const int kv = j * 64;
        asm volatile("s_wait_asynccnt 0x0" ::: "memory");
        __syncthreads();
        if (j + 1 < nb) stage(kv + 64, bufi ^ 1);

        // S = q @ k^T  (16 x 64): 4 key tiles x 2 K-chunks
        f32x8 st[4] = {};
#pragma unroll
        for (int g = 0; g < 4; ++g) {
            const char* kbase = (const char*)&lK[bufi][(g * 16 + (lane & 15)) * 64];
#pragma unroll
            for (int c = 0; c < 2; ++c) {
                bf16x16 kf = frag_cat(*(const bf16x8*)(kbase + c * 64 + bo),
                                      *(const bf16x8*)(kbase + c * 64 + bo + 32));
                st[g] = __builtin_amdgcn_wmma_f32_16x16x32_bf16(
                    false, qf[c], false, kf, (short)0, st[g], false, false);
            }
        }

        // causal mask + scale (1/sqrt(64)); row stats live in the 16-lane groups
#pragma unroll
        for (int r = 0; r < 8; ++r) {
            const int qrow = qw + r + ((lane & 16) ? 8 : 0);
#pragma unroll
            for (int g = 0; g < 4; ++g) {
                const int kcol = kv + g * 16 + (lane & 15);
                st[g][r] = (kcol <= qrow) ? st[g][r] * 0.125f : -3.0e38f;
            }
        }

        // row max: pure-VALU DPP butterfly per row (batched for ILP)
        float mx[8];
#pragma unroll
        for (int r = 0; r < 8; ++r)
            mx[r] = fmaxf(fmaxf(st[0][r], st[1][r]), fmaxf(st[2][r], st[3][r]));
#pragma unroll
        for (int r = 0; r < 8; ++r)
            mx[r] = dpp_max16(mx[r]);

        float sc[8];
#pragma unroll
        for (int r = 0; r < 8; ++r) {
            const float mn = fmaxf(m_r[r], mx[r]);
            sc[r]  = __expf(m_r[r] - mn);
            m_r[r] = mn;
        }
        // exponentiate probabilities
#pragma unroll
        for (int g = 0; g < 4; ++g)
#pragma unroll
            for (int r = 0; r < 8; ++r)
                st[g][r] = __expf(st[g][r] - m_r[r]);

        // P (C layout) -> LDS -> A-fragment layout
#pragma unroll
        for (int r = 0; r < 8; ++r) {
            const int prow = r + ((lane & 16) ? 8 : 0);
#pragma unroll
            for (int g = 0; g < 4; ++g)
                pb[wave][prow * 64 + g * 16 + (lane & 15)] = (bf16)st[g][r];
        }
        asm volatile("s_wait_dscnt 0x0" ::: "memory");

        const char* pbase = (const char*)&pb[wave][(lane & 15) * 64];
        bf16x16 pf[2];
#pragma unroll
        for (int c = 0; c < 2; ++c)
            pf[c] = frag_cat(*(const bf16x8*)(pbase + c * 64 + ao),
                             *(const bf16x8*)(pbase + c * 64 + ao + 32));

        // row sums via WMMA: rsv[r] = sum_k P[row][k]  (broadcast across the row group)
        f32x8 rsv = {};
#pragma unroll
        for (int c = 0; c < 2; ++c)
            rsv = __builtin_amdgcn_wmma_f32_16x16x32_bf16(
                false, pf[c], false, onesf, (short)0, rsv, false, false);

        // online-softmax state update + rescale accumulators
#pragma unroll
        for (int r = 0; r < 8; ++r) {
            l_r[r] = l_r[r] * sc[r] + rsv[r];
#pragma unroll
            for (int n = 0; n < 4; ++n) acc[n][r] *= sc[r];
        }

        // acc += P @ V  (contraction over 64 keys = 2 chunks, 4 dim tiles)
#pragma unroll
        for (int n = 0; n < 4; ++n) {
            const char* vbase = (const char*)&lV[bufi][(n * 16 + (lane & 15)) * 64];
#pragma unroll
            for (int c = 0; c < 2; ++c) {
                bf16x16 vf = frag_cat(*(const bf16x8*)(vbase + c * 64 + bo),
                                      *(const bf16x8*)(vbase + c * 64 + bo + 32));
                acc[n] = __builtin_amdgcn_wmma_f32_16x16x32_bf16(
                    false, pf[c], false, vf, (short)0, acc[n], false, false);
            }
        }
    }

    // normalize and store attn output as [B,T,H*D] bf16
#pragma unroll
    for (int r = 0; r < 8; ++r) {
        const float invl = __builtin_amdgcn_rcpf(l_r[r]);
        const int trow = qw + r + ((lane & 16) ? 8 : 0);
#pragma unroll
        for (int n = 0; n < 4; ++n) {
            const int col = h * Dd + n * 16 + (lane & 15);
            out[((size_t)b * Ts + trow) * (size_t)Cc + col] = (bf16)(acc[n][r] * invl);
        }
    }
}

// ---------------- launcher ----------------
extern "C" void kernel_launch(void* const* d_in, const int* in_sizes, int n_in,
                              void* d_out, int out_size, void* d_ws, size_t ws_size,
                              hipStream_t stream)
{
    (void)in_sizes; (void)n_in; (void)out_size; (void)ws_size;
    const float* x    = (const float*)d_in[0];
    const float* Wqkv = (const float*)d_in[1];
    const float* Wout = (const float*)d_in[2];
    float* outp = (float*)d_out;

    char* ws = (char*)d_ws;
    size_t off = 0;
    auto carve = [&](size_t bytes) -> char* {
        char* p = ws + off;
        off += (bytes + 255) & ~(size_t)255;
        return p;
    };
    bf16* xb   = (bf16*)carve((size_t)Mrows * Cc * 2);  // x in bf16
    bf16* wqt  = (bf16*)carve((size_t)N3 * Cc * 2);     // W_qkv^T bf16 [3072,1024]
    bf16* wot  = (bf16*)carve((size_t)Cc * Cc * 2);     // W_out^T bf16 [1024,1024]
    bf16* qkv  = (bf16*)carve((size_t)Mrows * N3 * 2);  // qkv bf16 [8192,3072]
    bf16* qrp  = (bf16*)carve((size_t)Mrows * Cc * 2);  // q rotated [B,H,T,D]
    bf16* krp  = (bf16*)carve((size_t)Mrows * Cc * 2);  // k rotated [B,H,T,D]
    bf16* vtp  = (bf16*)carve((size_t)Mrows * Cc * 2);  // v transposed [B,H,D,T]
    bf16* aout = (bf16*)carve((size_t)Mrows * Cc * 2);  // attention output [B,T,C]

    cvt_bf16_kernel<<<(Mrows * Cc) / 256, 256, 0, stream>>>(x, xb, Mrows * Cc);
    cvt_transpose_kernel<<<(Cc * N3) / 256, 256, 0, stream>>>(Wqkv, wqt, Cc, N3);
    cvt_transpose_kernel<<<(Cc * Cc) / 256, 256, 0, stream>>>(Wout, wot, Cc, Cc);

    gemm_bf16_kernel<bf16><<<dim3(N3 / 128, Mrows / 128), 256, 0, stream>>>(
        xb, wqt, qkv, Mrows, N3, Cc);

    rope_kernel<<<(Bb * Hh * Ts * 32) / 256, 256, 0, stream>>>(qkv, qrp, krp, vtp);

    attn_kernel<<<dim3(Bb * Hh, Ts / 64), 128, 0, stream>>>(qrp, krp, vtp, aout);

    gemm_bf16_kernel<float><<<dim3(Cc / 128, Mrows / 128), 256, 0, stream>>>(
        aout, wot, outp, Mrows, Cc, Cc);
}
